// CLUTRRV4_27144193310730
// MI455X (gfx1250) — matrix operand order repaired and checked by reading, not compiled, not run on hardware
//
#include <hip/hip_runtime.h>
#include <math.h>

// ---------------------------------------------------------------------------
// CLUTRR GNN for MI455X (gfx1250): bf16 WMMA (v_wmma_f32_16x16x32_bf16),
// one 128-thread block (4 wave32) per sample, all activations in LDS,
// weights pre-packed to bf16 fragments in d_ws (per-lane ISA B layout).
// ---------------------------------------------------------------------------

typedef __attribute__((ext_vector_type(16))) __bf16 bf16x16;
typedef __attribute__((ext_vector_type(8)))  __bf16 bf16x8;
typedef __attribute__((ext_vector_type(8)))  float  f32x8;

#define N_ENT 32
#define D     128
#define E     64
#define NSTEP 8

// LDS row strides (in bf16 elems), padded to keep 16B alignment + dodge bank conflicts
#define XS 392   // X buffer stride  (>= 384), 392*2 bytes % 16 == 0
#define HS 264   // H buffer stride  (>= 256), 264*2 bytes % 16 == 0

__device__ __forceinline__ float gelu_exact(float x) {
    return 0.5f * x * (1.0f + erff(x * 0.70710678118654752f));
}

// A fragment (16x32 bf16, M x K) from LDS, per CDNA5 ISA 16-bit A layout:
// lanes 0-15: M=lane, elems 0..7 -> K=kt*32+0..7,  elems 8..15 -> K=kt*32+16..23
// lanes16-31: M=lane-16, elems 0..7 -> K=kt*32+8..15, elems 8..15 -> K=kt*32+24..31
__device__ __forceinline__ bf16x16 load_a_lds(const __bf16* base, int stride,
                                              int mt, int kt, int lane) {
    const int m     = mt * 16 + (lane & 15);
    const int khalf = (lane >> 4) * 8;
    const __bf16* p = base + m * stride + kt * 32 + khalf;
    union { bf16x16 v; bf16x8 h[2]; } u;
    u.h[0] = *(const bf16x8*)(p);        // ds_load_b128
    u.h[1] = *(const bf16x8*)(p + 16);   // ds_load_b128
    return u.v;
}

// B fragment (32x16 bf16, K x N) from global, pre-packed so each lane's 16
// values are contiguous: packed[tile*512 + lane*16 + j] = W[K(l,j)][N(l)]
__device__ __forceinline__ bf16x16 load_b_glob(const __bf16* packed, int tile, int lane) {
    return *(const bf16x16*)(packed + ((size_t)tile << 9) + (lane << 4));
}

__device__ __forceinline__ f32x8 wmma_bf16(bf16x16 a, bf16x16 b, f32x8 c) {
    return __builtin_amdgcn_wmma_f32_16x16x32_bf16(false, a, false, b,
                                                   (short)0, c, false, false);
}

// ---------------------------------------------------------------------------
// Prologue: convert fp32 weight (K x N, row major) to bf16 fragments in the
// per-lane WMMA B layout. k = kt*32 + (l>>4)*8 + (j&7) + (j>>3)*16 ; n = nt*16 + (l&15)
// ---------------------------------------------------------------------------
__global__ void pack_weight_kernel(const float* __restrict__ W,
                                   __bf16* __restrict__ dst, int K, int N) {
    const int NT = N >> 4;
    int p = blockIdx.x * 256 + threadIdx.x;
    if (p >= K * N) return;
    int tile   = p >> 9;
    int within = p & 511;
    int l  = within >> 4;
    int j  = within & 15;
    int kt = tile / NT;
    int nt = tile - kt * NT;
    int k  = kt * 32 + ((l >> 4) << 3) + (j & 7) + ((j >> 3) << 4);
    int n  = nt * 16 + (l & 15);
    dst[p] = (__bf16)W[k * N + n];
}

// ---------------------------------------------------------------------------
// Main kernel: one block per sample.
// ---------------------------------------------------------------------------
__global__ __launch_bounds__(128)
void clutrr_main_kernel(const int* __restrict__ edge_src,
                        const int* __restrict__ edge_rel,
                        const int* __restrict__ edge_tgt,
                        const int* __restrict__ n_edges,
                        const int* __restrict__ query_src,
                        const int* __restrict__ query_tgt,
                        const float* __restrict__ entity_embed,
                        const float* __restrict__ rel_embed,
                        const float* __restrict__ msg_b1,
                        const float* __restrict__ msg_b2,
                        const float* __restrict__ upd_b1,
                        const float* __restrict__ upd_b2,
                        const float* __restrict__ cls_W1,
                        const float* __restrict__ cls_b1,
                        const float* __restrict__ cls_W2,
                        const float* __restrict__ cls_b2,
                        const __bf16* __restrict__ pW1,
                        const __bf16* __restrict__ pW2,
                        const __bf16* __restrict__ pU1,
                        const __bf16* __restrict__ pU2,
                        float* __restrict__ out) {
    __shared__ float   S[N_ENT * D];       // 16 KB  fp32 entity states
    __shared__ float   AGG[N_ENT * D];     // 16 KB  fp32 scatter-add target
    __shared__ __bf16  RELB[E * D];        // 16 KB  gathered relation embeds
    __shared__ __bf16  XB[E * XS];         // 49 KB  GEMM input buffer
    __shared__ __bf16  HB[E * HS];         // 33 KB  GEMM hidden buffer
    __shared__ int     SRCI[E];
    __shared__ int     TGTI[E];

    const int b    = blockIdx.x;
    const int tid  = threadIdx.x;
    const int lane = tid & 31;
    const int wave = tid >> 5;
    const int cn   = lane & 15;        // column within 16-wide tile
    const int r0   = (lane >> 4) * 8;  // row base within 16-tall tile (C layout)

    // ---- init: S from shared table, edge indices, gathered rel embeds ----
    for (int i = tid; i < N_ENT * D; i += 128) S[i] = entity_embed[i];
    if (tid < E) {
        SRCI[tid] = edge_src[b * E + tid];
        TGTI[tid] = edge_tgt[b * E + tid];
    }
    for (int i = tid; i < E * D; i += 128) {
        int e = i >> 7, c = i & 127;
        RELB[i] = (__bf16)rel_embed[edge_rel[b * E + e] * D + c];
    }
    const int nE = n_edges[b];
    __syncthreads();

    for (int step = 0; step < NSTEP; ++step) {
        // ---- Phase 1: X1 = [S[src] | rel | S[tgt]]  (64 x 384), zero AGG ----
        for (int i = tid; i < E * 384; i += 128) {
            int e = i / 384, c = i - e * 384;
            __bf16 v;
            if (c < 128)      v = (__bf16)S[SRCI[e] * D + c];
            else if (c < 256) v = RELB[e * D + (c - 128)];
            else              v = (__bf16)S[TGTI[e] * D + (c - 256)];
            XB[e * XS + c] = v;
        }
        for (int i = tid; i < N_ENT * D; i += 128) AGG[i] = 0.0f;
        __syncthreads();

        // ---- Phase 2: H = gelu(X1 @ msg_W1 + b1)   64x384 @ 384x256 ----
        // tiles: 4(M) x 16(N); wave w owns nt in [4w, 4w+3]
        for (int t = 0; t < 16; ++t) {
            const int nt = wave * 4 + (t & 3);
            const int mt = t >> 2;
            const float bias = msg_b1[nt * 16 + cn];
            f32x8 acc;
#pragma unroll
            for (int r = 0; r < 8; ++r) acc[r] = bias;
#pragma unroll
            for (int kt = 0; kt < 12; ++kt) {
                bf16x16 a  = load_a_lds(XB, XS, mt, kt, lane);
                bf16x16 bb = load_b_glob(pW1, kt * 16 + nt, lane);
                acc = wmma_bf16(a, bb, acc);
            }
            const int n = nt * 16 + cn;
#pragma unroll
            for (int r = 0; r < 8; ++r)
                HB[(mt * 16 + r0 + r) * HS + n] = (__bf16)gelu_exact(acc[r]);
        }
        __syncthreads();

        // ---- Phase 3: messages = H @ msg_W2 + b2; mask; scatter-add ----
        // 64x256 @ 256x128, tiles 4(M) x 8(N); wave w owns nt in {2w, 2w+1}
        for (int t = 0; t < 8; ++t) {
            const int nt = wave * 2 + (t & 1);
            const int mt = t >> 1;
            const float bias = msg_b2[nt * 16 + cn];
            f32x8 acc;
#pragma unroll
            for (int r = 0; r < 8; ++r) acc[r] = bias;
#pragma unroll
            for (int kt = 0; kt < 8; ++kt) {
                bf16x16 a  = load_a_lds(HB, HS, mt, kt, lane);
                bf16x16 bb = load_b_glob(pW2, kt * 8 + nt, lane);
                acc = wmma_bf16(a, bb, acc);
            }
            const int n = nt * 16 + cn;
#pragma unroll
            for (int r = 0; r < 8; ++r) {
                const int e = mt * 16 + r0 + r;
                if (e < nE) atomicAdd(&AGG[TGTI[e] * D + n], acc[r]);  // ds_add_f32
            }
        }
        __syncthreads();

        // ---- Phase 4: X2 = [S | AGG]  (32 x 256) ----
        for (int i = tid; i < N_ENT * 256; i += 128) {
            int m = i >> 8, c = i & 255;
            float v = (c < 128) ? S[m * D + c] : AGG[m * D + (c - 128)];
            XB[m * XS + c] = (__bf16)v;
        }
        __syncthreads();

        // ---- Phase 5: H2 = gelu(X2 @ upd_W1 + b1)  32x256 @ 256x256 ----
        // tiles 2(M) x 16(N); wave w owns nt in [4w, 4w+3]
        for (int t = 0; t < 8; ++t) {
            const int nt = wave * 4 + (t & 3);
            const int mt = t >> 2;
            const float bias = upd_b1[nt * 16 + cn];
            f32x8 acc;
#pragma unroll
            for (int r = 0; r < 8; ++r) acc[r] = bias;
#pragma unroll
            for (int kt = 0; kt < 8; ++kt) {
                bf16x16 a  = load_a_lds(XB, XS, mt, kt, lane);
                bf16x16 bb = load_b_glob(pU1, kt * 16 + nt, lane);
                acc = wmma_bf16(a, bb, acc);
            }
            const int n = nt * 16 + cn;
#pragma unroll
            for (int r = 0; r < 8; ++r)
                HB[(mt * 16 + r0 + r) * HS + n] = (__bf16)gelu_exact(acc[r]);
        }
        __syncthreads();

        // ---- Phase 6: S += H2 @ upd_W2 + b2  32x256 @ 256x128 ----
        // tiles 2(M) x 8(N); wave w owns nt in {2w, 2w+1}
        for (int t = 0; t < 4; ++t) {
            const int nt = wave * 2 + (t & 1);
            const int mt = t >> 1;
            const float bias = upd_b2[nt * 16 + cn];
            f32x8 acc;
#pragma unroll
            for (int r = 0; r < 8; ++r) acc[r] = bias;
#pragma unroll
            for (int kt = 0; kt < 8; ++kt) {
                bf16x16 a  = load_a_lds(HB, HS, mt, kt, lane);
                bf16x16 bb = load_b_glob(pU2, kt * 8 + nt, lane);
                acc = wmma_bf16(a, bb, acc);
            }
            const int n = nt * 16 + cn;
#pragma unroll
            for (int r = 0; r < 8; ++r)
                S[(mt * 16 + r0 + r) * D + n] += acc[r];   // unique (m,n) per lane
        }
        __syncthreads();
    }

    // ---- Classifier head: h = gelu([S[qs]|S[qt]] @ cls_W1 + b1); out = h @ cls_W2 + b2
    const int qs = query_src[b], qt = query_tgt[b];
    float* hbuf = (float*)XB;  // reuse LDS
    {
        float acc = cls_b1[tid];
        for (int k = 0; k < 128; ++k) acc += S[qs * D + k] * cls_W1[k * 128 + tid];
        for (int k = 0; k < 128; ++k) acc += S[qt * D + k] * cls_W1[(128 + k) * 128 + tid];
        hbuf[tid] = gelu_exact(acc);
    }
    __syncthreads();
    if (tid < 20) {
        float acc = cls_b2[tid];
        for (int k = 0; k < 128; ++k) acc += hbuf[k] * cls_W2[k * 20 + tid];
        out[b * 20 + tid] = acc;
    }
}

// ---------------------------------------------------------------------------
extern "C" void kernel_launch(void* const* d_in, const int* in_sizes, int n_in,
                              void* d_out, int out_size, void* d_ws, size_t ws_size,
                              hipStream_t stream) {
    const int*   edge_src     = (const int*)d_in[0];
    const int*   edge_rel     = (const int*)d_in[1];
    const int*   edge_tgt     = (const int*)d_in[2];
    const int*   n_edges      = (const int*)d_in[3];
    const int*   query_src    = (const int*)d_in[4];
    const int*   query_tgt    = (const int*)d_in[5];
    const float* entity_embed = (const float*)d_in[6];
    const float* rel_embed    = (const float*)d_in[7];
    const float* msg_W1       = (const float*)d_in[8];
    const float* msg_b1       = (const float*)d_in[9];
    const float* msg_W2       = (const float*)d_in[10];
    const float* msg_b2       = (const float*)d_in[11];
    const float* upd_W1       = (const float*)d_in[12];
    const float* upd_b1       = (const float*)d_in[13];
    const float* upd_W2       = (const float*)d_in[14];
    const float* upd_b2       = (const float*)d_in[15];
    const float* cls_W1       = (const float*)d_in[16];
    const float* cls_b1       = (const float*)d_in[17];
    const float* cls_W2       = (const float*)d_in[18];
    const float* cls_b2       = (const float*)d_in[19];

    const int B = in_sizes[3];  // 2048

    // workspace carve: packed bf16 weight fragments (~448 KB total)
    __bf16* pW1 = (__bf16*)d_ws;            // 384x256 = 98304
    __bf16* pW2 = pW1 + 384 * 256;          // 256x128 = 32768
    __bf16* pU1 = pW2 + 256 * 128;          // 256x256 = 65536
    __bf16* pU2 = pU1 + 256 * 256;          // 256x128 = 32768

    pack_weight_kernel<<<(384 * 256 + 255) / 256, 256, 0, stream>>>(msg_W1, pW1, 384, 256);
    pack_weight_kernel<<<(256 * 128 + 255) / 256, 256, 0, stream>>>(msg_W2, pW2, 256, 128);
    pack_weight_kernel<<<(256 * 256 + 255) / 256, 256, 0, stream>>>(upd_W1, pU1, 256, 256);
    pack_weight_kernel<<<(256 * 128 + 255) / 256, 256, 0, stream>>>(upd_W2, pU2, 256, 128);

    clutrr_main_kernel<<<B, 128, 0, stream>>>(
        edge_src, edge_rel, edge_tgt, n_edges, query_src, query_tgt,
        entity_embed, rel_embed, msg_b1, msg_b2, upd_b1, upd_b2,
        cls_W1, cls_b1, cls_W2, cls_b2,
        pW1, pW2, pU1, pU2, (float*)d_out);
}
